// TemporalFrameContrastiveLoss_72954314490493
// MI455X (gfx1250) — matrix operand order repaired and checked by reading, not compiled, not run on hardware
//
#include <hip/hip_runtime.h>
#include <hip/hip_bf16.h>

typedef __attribute__((ext_vector_type(16))) _Float16 v16h;
typedef __attribute__((ext_vector_type(8)))  _Float16 v8h;
typedef __attribute__((ext_vector_type(4)))  _Float16 v4h;
typedef __attribute__((ext_vector_type(8)))  float    v8f;
typedef __attribute__((ext_vector_type(4)))  float    v4f;

#define B_ROWS   8192
#define DIM      128
#define TILE     16
#define NT       (B_ROWS / TILE)   // 512 tiles per dimension
#define RT       2                 // row tiles per wave job
#define CT       4                 // column tiles per wave job
#define NRG      (NT / RT)         // 256 row groups
#define NCG      (NT / CT)         // 128 column groups
#define NTRK     256               // track-id bins (NUM_TRACKS)
#define NCHUNK   64                // sort chunks
#define CHROWS   (B_ROWS / NCHUNK) // 128 rows per chunk
#define ACC_POS_SUM 0
#define ACC_POS_CNT 1
#define ACC_NEG_SUM 2
#define ACC_NEG_CNT 3

// workspace byte offsets
#define WS_ACC    0        // 4 floats
#define WS_BAND   64       // 1 int
#define WS_TBASE  256      // 256 ints   (track base, unused outside scan but kept)
#define WS_HIST   2048     // 64*256 ints: per-chunk hist -> absolute chunk starts
#define WS_POS    69632    // 8192 ints: row -> permuted position
#define WS_PF     102400   // 8192 floats: permuted frame values
#define WS_PG     135168   // 8192 ints: permuted track ids
#define WS_NH     262144   // 8192*128 f16: permuted normalized matrix (2 MB)

static __device__ __forceinline__ v16h combine16(v8h lo, v8h hi) {
    v16h r;
#pragma unroll
    for (int i = 0; i < 8; ++i) { r[i] = lo[i]; r[i + 8] = hi[i]; }
    return r;
}

// ---------------------------------------------------------------------------
// Kernel 1: per-chunk track histograms (64 blocks); block 0 zeroes accumulators
// ---------------------------------------------------------------------------
__global__ __launch_bounds__(256) void tfc_hist_kernel(
    const int* __restrict__ gids, int* __restrict__ hist,
    float* __restrict__ acc) {
    __shared__ int h[NTRK];
    h[threadIdx.x] = 0;
    if (blockIdx.x == 0 && threadIdx.x < 4) acc[threadIdx.x] = 0.0f;
    __syncthreads();
    const int c = (int)blockIdx.x;
    if (threadIdx.x < CHROWS) {
        const int g = gids[c * CHROWS + threadIdx.x] & (NTRK - 1);
        atomicAdd(&h[g], 1);
    }
    __syncthreads();
    hist[c * NTRK + threadIdx.x] = h[threadIdx.x];
}

// ---------------------------------------------------------------------------
// Kernel 2 (1 block, 256 threads): thread t owns track t.
//  - convert hist[c][t] to absolute permuted start offsets (stable sort bases)
//  - band = max track population (max same-track index distance bound)
// ---------------------------------------------------------------------------
__global__ __launch_bounds__(256) void tfc_scan_kernel(
    int* __restrict__ hist, int* __restrict__ tbase, int* __restrict__ band) {
    const int t = (int)threadIdx.x;
    int run = 0;
#pragma unroll 4
    for (int c = 0; c < NCHUNK; ++c) {      // per-track prefix over chunks
        const int v = hist[c * NTRK + t];
        hist[c * NTRK + t] = run;
        run += v;
    }
    __shared__ int tot[NTRK];
    __shared__ int scan[NTRK];
    __shared__ int mx[NTRK];
    tot[t] = run; scan[t] = run; mx[t] = run;
    __syncthreads();
    for (int off = 1; off < NTRK; off <<= 1) {   // inclusive scan over tracks
        const int v = (t >= off) ? scan[t - off] : 0;
        __syncthreads();
        scan[t] += v;
        __syncthreads();
    }
    const int base = scan[t] - tot[t];           // exclusive: track base
    tbase[t] = base;
#pragma unroll 4
    for (int c = 0; c < NCHUNK; ++c) hist[c * NTRK + t] += base;
    for (int off = NTRK / 2; off >= 1; off >>= 1) {
        if (t < off) mx[t] = max(mx[t], mx[t + off]);
        __syncthreads();
    }
    if (t == 0) band[0] = mx[0];
}

// ---------------------------------------------------------------------------
// Kernel 3: stable scatter (64 blocks). Thread t owns track t, walks its chunk
// in row order -> deterministic permutation. pos[row] = permuted position.
// ---------------------------------------------------------------------------
__global__ __launch_bounds__(256) void tfc_scatter_kernel(
    const int* __restrict__ gids, const int* __restrict__ hist,
    int* __restrict__ pos) {
    const int t = (int)threadIdx.x;
    const int c = (int)blockIdx.x;
    int local = hist[c * NTRK + t];
    for (int i = 0; i < CHROWS; ++i) {
        const int row = c * CHROWS + i;
        const int g = gids[row] & (NTRK - 1);
        if (g == t) pos[row] = local++;
    }
}

// ---------------------------------------------------------------------------
// Kernel 4: L2-normalize rows (fp32 -> f16), written to permuted position;
// also emits permuted frame (as float) and track-id arrays.
// One wave per row; 4 floats per lane.
// ---------------------------------------------------------------------------
__global__ __launch_bounds__(256) void tfc_normalize_kernel(
    const float* __restrict__ emb, const int* __restrict__ frames,
    const int* __restrict__ gids, const int* __restrict__ pos,
    _Float16* __restrict__ nh, float* __restrict__ pf, int* __restrict__ pg) {
    const int row  = (int)((blockIdx.x * blockDim.x + threadIdx.x) >> 5);
    const int lane = (int)(threadIdx.x & 31);
    if (row >= B_ROWS) return;

    const v4f x = ((const v4f*)(emb + (size_t)row * DIM))[lane];
    float ss = x[0] * x[0] + x[1] * x[1] + x[2] * x[2] + x[3] * x[3];
#pragma unroll
    for (int m = 16; m >= 1; m >>= 1) ss += __shfl_xor(ss, m, 32);

    const float inv = 1.0f / fmaxf(sqrtf(ss), 1e-6f);
    v4h h;
    h[0] = (_Float16)(x[0] * inv);
    h[1] = (_Float16)(x[1] * inv);
    h[2] = (_Float16)(x[2] * inv);
    h[3] = (_Float16)(x[3] * inv);
    const int p = pos[row];
    ((v4h*)(nh + (size_t)p * DIM))[lane] = h;
    if (lane == 0) {
        pf[p] = (float)frames[row];
        pg[p] = gids[row];
    }
}

// ---------------------------------------------------------------------------
// Kernel 5: banded Gram via V_WMMA_F32_16X16X32_F16 + fused masked reduction.
// After the track sort, same-track pairs satisfy q - p < band, so any job
// whose minimum (q - p) >= band can exit immediately -> O(B * band) pairs.
//
// Operand k order used here (contiguous, not the ISA table order; wave32):
//   A (16x32 f16):  lane L<16 holds row M=L,    K = [0..15]
//                   lane L>=16 holds row M=L-16, K = [16..31]
//   B (32x16 f16):  lane L<16 holds col N=L,    K = [0..15]
//                   lane L>=16 holds col N=L-16, K = [16..31]
//   B = n^T -> column j of B is row j of n: contiguous b128 loads.
//   C/D (16x16 f32): element v of v8f: M = v + 8*(lane>=16), N = lane&15.
// ---------------------------------------------------------------------------
__global__ __launch_bounds__(256) void tfc_gram_kernel(
    const _Float16* __restrict__ nh,
    const float* __restrict__ pf,
    const int* __restrict__ pg,
    const int* __restrict__ span_p,
    const int* __restrict__ band_p,
    float* __restrict__ acc) {
    const int lane = (int)(threadIdx.x & 31);
    const int job  = (int)((blockIdx.x * blockDim.x + threadIdx.x) >> 5);
    const int rg   = job / NCG;
    const int cg   = job % NCG;
    const int tr0  = rg * RT;
    const int tc0  = cg * CT;
    if (tc0 + CT - 1 < tr0) return;               // below diagonal (uniform)
    const int band = band_p[0];                   // max track size
    // min (q - p) in this job block = tc0*16 - ((tr0+RT)*16 - 1)
    if ((tc0 - tr0 - RT) * TILE + 1 >= band) return;   // outside band (uniform)

    const float neg_min = 0.5f * (float)span_p[0];

    const int l15  = lane & 15;
    const int hsel = lane >> 4;

    v8f c[RT][CT];
#pragma unroll
    for (int r = 0; r < RT; ++r)
#pragma unroll
        for (int t = 0; t < CT; ++t) { v8f z = {}; c[r][t] = z; }

#pragma unroll
    for (int k = 0; k < DIM / 32; ++k) {          // 4 K-chunks of 32 halves
        const int kb = k * 32;
        v16h a[RT];
#pragma unroll
        for (int r = 0; r < RT; ++r) {
            const int arow = (tr0 + r) * TILE + l15;
            const _Float16* ap = nh + (size_t)arow * DIM + kb;
            const v8h a_lo = *(const v8h*)(ap + hsel * 16);
            const v8h a_hi = *(const v8h*)(ap + 8 + hsel * 16);
            a[r] = combine16(a_lo, a_hi);
        }
#pragma unroll
        for (int t = 0; t < CT; ++t) {
            const int brow = (tc0 + t) * TILE + l15;   // B col = n row
            const _Float16* bp = nh + (size_t)brow * DIM + kb + hsel * 16;
            const v8h b_lo = *(const v8h*)(bp);
            const v8h b_hi = *(const v8h*)(bp + 8);
            const v16h b = combine16(b_lo, b_hi);
#pragma unroll
            for (int r = 0; r < RT; ++r) {
                c[r][t] = __builtin_amdgcn_wmma_f32_16x16x32_f16(
                    false, a[r], false, b, (short)0, c[r][t], false, false);
            }
        }
    }

    // Preload the RT x 8 (M-row) frame/track values this lane tests.
    float fi[RT][8]; int gi[RT][8];
#pragma unroll
    for (int r = 0; r < RT; ++r) {
        const int ib = (tr0 + r) * TILE + (hsel << 3);
#pragma unroll
        for (int v = 0; v < 8; ++v) {
            fi[r][v] = pf[ib + v];
            gi[r][v] = pg[ib + v];
        }
    }

    float ps = 0.0f, pc = 0.0f, ns = 0.0f, nc = 0.0f;
#pragma unroll
    for (int t = 0; t < CT; ++t) {
        const int tc = tc0 + t;
        const int   j  = tc * TILE + l15;
        const float fj = pf[j];
        const int   gj = pg[j];
#pragma unroll
        for (int r = 0; r < RT; ++r) {
            if (tc < tr0 + r) continue;           // subtile below diagonal
            const int ib = (tr0 + r) * TILE + (hsel << 3);
#pragma unroll
            for (int v = 0; v < 8; ++v) {
                const int i = ib + v;
                if (i < j && gi[r][v] == gj) {
                    const float fd = fabsf(fi[r][v] - fj);
                    const float s  = c[r][t][v];
                    if (fd <= 2.0f)   { ps += 1.0f - s;       pc += 1.0f; }
                    if (fd > neg_min) { ns += fmaxf(s, 0.0f); nc += 1.0f; }
                }
            }
        }
    }

#pragma unroll
    for (int m = 16; m >= 1; m >>= 1) {
        ps += __shfl_xor(ps, m, 32);
        pc += __shfl_xor(pc, m, 32);
        ns += __shfl_xor(ns, m, 32);
        nc += __shfl_xor(nc, m, 32);
    }
    if (lane == 0) {
        atomicAdd(acc + ACC_POS_SUM, ps);
        atomicAdd(acc + ACC_POS_CNT, pc);
        atomicAdd(acc + ACC_NEG_SUM, ns);
        atomicAdd(acc + ACC_NEG_CNT, nc);
    }
}

// ---------------------------------------------------------------------------
// Kernel 6: combine the four scalars into the loss.
// ---------------------------------------------------------------------------
__global__ void tfc_finalize_kernel(const float* __restrict__ acc,
                                    float* __restrict__ out) {
    if (threadIdx.x != 0 || blockIdx.x != 0) return;
    const float ps = acc[ACC_POS_SUM], pc = acc[ACC_POS_CNT];
    const float ns = acc[ACC_NEG_SUM], nc = acc[ACC_NEG_CNT];
    float loss = 0.0f;
    if (pc > 0.0f) loss += ps / fmaxf(pc, 1.0f);
    if (nc > 0.0f) loss += ns / fmaxf(nc, 1.0f);
    if (!(loss == loss) || loss == __builtin_inff() || loss == -__builtin_inff())
        loss = 0.0f;
    out[0] = loss;
}

extern "C" void kernel_launch(void* const* d_in, const int* in_sizes, int n_in,
                              void* d_out, int out_size, void* d_ws, size_t ws_size,
                              hipStream_t stream) {
    (void)in_sizes; (void)n_in; (void)out_size; (void)ws_size;
    const float* emb    = (const float*)d_in[0];   // [8192,128] f32
    const int*   frames = (const int*)d_in[1];     // [8192] i32
    const int*   gids   = (const int*)d_in[2];     // [8192] i32
    const int*   span   = (const int*)d_in[3];     // scalar i32

    char* ws = (char*)d_ws;
    float*    acc   = (float*)(ws + WS_ACC);
    int*      band  = (int*)(ws + WS_BAND);
    int*      tbase = (int*)(ws + WS_TBASE);
    int*      hist  = (int*)(ws + WS_HIST);
    int*      pos   = (int*)(ws + WS_POS);
    float*    pf    = (float*)(ws + WS_PF);
    int*      pg    = (int*)(ws + WS_PG);
    _Float16* nh    = (_Float16*)(ws + WS_NH);

    // 1) per-chunk histograms (+ zero accumulators)
    tfc_hist_kernel<<<NCHUNK, 256, 0, stream>>>(gids, hist, acc);
    // 2) scan -> absolute stable-sort offsets, band = max track size
    tfc_scan_kernel<<<1, 256, 0, stream>>>(hist, tbase, band);
    // 3) stable scatter -> pos[row]
    tfc_scatter_kernel<<<NCHUNK, 256, 0, stream>>>(gids, hist, pos);
    // 4) normalize into permuted order (8 waves/block -> 1024 blocks)
    tfc_normalize_kernel<<<B_ROWS / 8, 256, 0, stream>>>(emb, frames, gids, pos,
                                                         nh, pf, pg);
    // 5) banded gram + fused masked reduction over full job grid
    //    (32768 wave jobs, ~2% survive the band test; 8 waves/block)
    tfc_gram_kernel<<<(NRG * NCG) / 8, 256, 0, stream>>>(nh, pf, pg, span, band, acc);
    // 6) finalize scalar
    tfc_finalize_kernel<<<1, 32, 0, stream>>>(acc, (float*)d_out);
}
